// DKFN_10711648436450
// MI455X (gfx1250) — matrix operand order repaired and checked by reading, not compiled, not run on hardware
//
#include <hip/hip_runtime.h>
#include <cstdint>
#include <cstddef>

// ---------------------------------------------------------------------------
// Problem constants (from reference): B=8192, N=1024, K_HOPS=3
// ---------------------------------------------------------------------------
#define BATCH 8192
#define NDIM  1024
#define KHOP  3
#define CN    (KHOP * NDIM + NDIM)   // 4096 : combined feature width
#define RN    (2 * NDIM)             // 2048 : rcombined feature width

typedef __attribute__((ext_vector_type(16))) __bf16 v16bf;
typedef __attribute__((ext_vector_type(8)))  __bf16 bf16x8;
typedef __attribute__((ext_vector_type(8)))  float  v8f;

// ---------------------------------------------------------------------------
// WMMA helpers: bf16 16x16x32, f32 accumulate (CDNA5 VGPR layouts, wave32)
// ---------------------------------------------------------------------------
__device__ __forceinline__ v8f wmma_bf16(v16bf a, v16bf b, v8f c) {
    // 8 args: (neg_a, A, neg_b, B, c_mod, C, reuse_a, reuse_b)
    return __builtin_amdgcn_wmma_f32_16x16x32_bf16(false, a, false, b,
                                                   (short)0, c, false, false);
}

// A fragment: 16(M) x 32(K) bf16 from row-major matrix. `base` points at
// (m0,k0). Lane L: row = L&15; lanes 0-15 hold K={g..g+7, 16+g..16+g+7} with
// g=0, lanes 16-31 with g=8 (ISA 7.12.2, 16-bit A-matrix table).
__device__ __forceinline__ v16bf load_a_frag(const __bf16* base, int lda, int lane) {
    int row = lane & 15;
    int g   = (lane >> 4) << 3;              // 0 or 8
    const __bf16* p = base + (size_t)row * lda + g;
    bf16x8 lo = *(const bf16x8*)(p);         // K = g .. g+7
    bf16x8 hi = *(const bf16x8*)(p + 16);    // K = 16+g .. 16+g+7
    v16bf a;
#pragma unroll
    for (int e = 0; e < 8; ++e) { a[e] = lo[e]; a[e + 8] = hi[e]; }
    return a;
}

// B fragment: 32(K) x 16(N) bf16 from a row-major [Nout x Kdim] matrix
// (weights are already [n,k]); element [k][n] = W[n][k]. Lane L: col = L&15,
// lanes 0-15 carry K={0..7,16..23}, lanes 16-31 K={8..15,24..31}.
__device__ __forceinline__ v16bf load_b_frag(const __bf16* base, int ldb, int lane) {
    int col = lane & 15;
    int kb  = (lane >> 4) << 3;              // 0 or 8
    const __bf16* p = base + (size_t)col * ldb + kb;
    bf16x8 lo = *(const bf16x8*)(p);
    bf16x8 hi = *(const bf16x8*)(p + 16);
    v16bf b;
#pragma unroll
    for (int e = 0; e < 8; ++e) { b[e] = lo[e]; b[e + 8] = hi[e]; }
    return b;
}

__device__ __forceinline__ float sigm(float x) { return 1.0f / (1.0f + expf(-x)); }

#define ZERO8 {0.f,0.f,0.f,0.f,0.f,0.f,0.f,0.f}

// ---------------------------------------------------------------------------
// Small prep kernels
// ---------------------------------------------------------------------------
__global__ void convert_f32_bf16_kernel(const float* __restrict__ src,
                                        __bf16* __restrict__ dst, int n) {
    for (int i = blockIdx.x * blockDim.x + threadIdx.x; i < n;
         i += gridDim.x * blockDim.x)
        dst[i] = (__bf16)src[i];
}

// cs[j] = sum_i A[i][j]
__global__ void colsum_kernel(const float* __restrict__ A, float* __restrict__ cs) {
    int j = blockIdx.x * blockDim.x + threadIdx.x;
    if (j < NDIM) {
        float s = 0.f;
        for (int i = 0; i < NDIM; ++i) s += A[(size_t)i * NDIM + j];
        cs[j] = s;
    }
}

// normA[i][j] = A[i][j]/cs[i]; emit normA^T (bf16, B-operand form) and
// A1 = min(normA, 1) in f32 + bf16.
__global__ void build_a1_kernel(const float* __restrict__ A,
                                const float* __restrict__ cs,
                                __bf16* __restrict__ normAT,
                                float* __restrict__ Af0,
                                __bf16* __restrict__ Abf0) {
    int idx = blockIdx.x * blockDim.x + threadIdx.x;   // exactly N*N threads
    int i = idx >> 10, j = idx & (NDIM - 1);
    float v = A[idx] / cs[i];
    normAT[(size_t)j * NDIM + i] = (__bf16)v;
    float a1 = fminf(v, 1.0f);
    Af0[idx]  = a1;
    Abf0[idx] = (__bf16)a1;
}

// filt[k][m][n] = A_k[m][n] * W_gc[k][m][n]  (bf16)
__global__ void filt_kernel(const float* __restrict__ Af,
                            const float* __restrict__ Wgc,
                            __bf16* __restrict__ filt) {
    int idx = blockIdx.x * blockDim.x + threadIdx.x;   // 3*N*N threads
    filt[idx] = (__bf16)(Af[idx] * Wgc[idx]);
}

// nw2[m] = sum_n A3[m][n] * nw[n]
__global__ void nw2_kernel(const float* __restrict__ A3,
                           const float* __restrict__ nw,
                           float* __restrict__ nw2) {
    __shared__ float sm[256];
    int m = blockIdx.x, tid = threadIdx.x;
    float s = 0.f;
    for (int n = tid; n < NDIM; n += 256) s += A3[(size_t)m * NDIM + n] * nw[n];
    sm[tid] = s; __syncthreads();
    for (int o = 128; o > 0; o >>= 1) {
        if (tid < o) sm[tid] += sm[tid + o];
        __syncthreads();
    }
    if (tid == 0) nw2[m] = sm[0];
}

// rcombined = [input | rHidden] (bf16);  combined[:,3072:4096] = Hidden (bf16)
__global__ void pack_kernel(const float* __restrict__ input,
                            const float* __restrict__ rhid,
                            const float* __restrict__ hid,
                            __bf16* __restrict__ rcomb,
                            __bf16* __restrict__ comb) {
    int idx = blockIdx.x * blockDim.x + threadIdx.x;   // BATCH*N threads
    int b = idx >> 10, n = idx & (NDIM - 1);
    rcomb[(size_t)b * RN + n]         = (__bf16)input[idx];
    rcomb[(size_t)b * RN + NDIM + n]  = (__bf16)rhid[idx];
    comb [(size_t)b * CN + KHOP * NDIM + n] = (__bf16)hid[idx];
}

// ---------------------------------------------------------------------------
// GEMM kernels (256 threads = 8 waves per block)
// ---------------------------------------------------------------------------

// A_next = min(A_prev @ normA, 1):  C[i][j] = sum_k Aprev[i][k]*normAT[j][k]
// (tiny: 2x (1024^3); one 16x16 tile per wave)
__global__ void chain_gemm_kernel(const __bf16* __restrict__ Abf_prev,
                                  const __bf16* __restrict__ normAT,
                                  float* __restrict__ Af_next,
                                  __bf16* __restrict__ Abf_next) {
    int lane = threadIdx.x & 31, wave = threadIdx.x >> 5;
    int m0 = (blockIdx.y * 8 + wave) * 16;
    int n0 = blockIdx.x * 16;
    v8f acc = ZERO8;
    for (int k0 = 0; k0 < NDIM; k0 += 32) {
        v16bf a = load_a_frag(Abf_prev + (size_t)m0 * NDIM + k0, NDIM, lane);
        v16bf b = load_b_frag(normAT   + (size_t)n0 * NDIM + k0, NDIM, lane);
        acc = wmma_bf16(a, b, acc);
    }
    int col = n0 + (lane & 15), mb = (lane >> 4) * 8;
#pragma unroll
    for (int r = 0; r < 8; ++r) {
        size_t idx = (size_t)(m0 + mb + r) * NDIM + col;
        float v = fminf(acc[r], 1.0f);
        Af_next[idx]  = v;
        Abf_next[idx] = (__bf16)v;
    }
}

// gc_k = input @ filt_k^T : C[b][m] = sum_n input[b][n]*filt_k[m][n]
// 2(M) x 2(N) register blocking: each wave owns a 32x32 C region.
// input rows live in rcomb[:, 0:1024] (lda = RN). Writes f32 gc and bf16
// combined[:, k*N : (k+1)*N].
__global__ void gc_gemm_kernel(const __bf16* __restrict__ rcomb,
                               const __bf16* __restrict__ filt,
                               float* __restrict__ gc_out,
                               __bf16* __restrict__ comb) {
    int lane = threadIdx.x & 31, wave = threadIdx.x >> 5;
    int m0 = (blockIdx.y * 8 + wave) * 32;       // batch rows (2 tiles)
    int n0 = blockIdx.x * 32;                    // output features m (2 tiles)
    int k  = blockIdx.z;                         // hop
    const __bf16* fk = filt + (size_t)k * NDIM * NDIM;
    v8f acc[2][2] = {{ZERO8, ZERO8}, {ZERO8, ZERO8}};
    for (int k0 = 0; k0 < NDIM; k0 += 32) {
        v16bf a0 = load_a_frag(rcomb + (size_t)(m0)      * RN + k0, RN, lane);
        v16bf a1 = load_a_frag(rcomb + (size_t)(m0 + 16) * RN + k0, RN, lane);
        v16bf b0 = load_b_frag(fk + (size_t)(n0)      * NDIM + k0, NDIM, lane);
        v16bf b1 = load_b_frag(fk + (size_t)(n0 + 16) * NDIM + k0, NDIM, lane);
        acc[0][0] = wmma_bf16(a0, b0, acc[0][0]);
        acc[0][1] = wmma_bf16(a0, b1, acc[0][1]);
        acc[1][0] = wmma_bf16(a1, b0, acc[1][0]);
        acc[1][1] = wmma_bf16(a1, b1, acc[1][1]);
    }
    int mb = (lane >> 4) * 8;
#pragma unroll
    for (int mt = 0; mt < 2; ++mt) {
#pragma unroll
        for (int nt = 0; nt < 2; ++nt) {
            int col = n0 + nt * 16 + (lane & 15);
#pragma unroll
            for (int r = 0; r < 8; ++r) {
                int row = m0 + mt * 16 + mb + r;
                float v = acc[mt][nt][r];
                gc_out[(size_t)row * (KHOP * NDIM) + (size_t)k * NDIM + col] = v;
                comb  [(size_t)row * CN            + (size_t)k * NDIM + col] = (__bf16)v;
            }
        }
    }
}

// Four gate GEMMs fused with the LSTM cell update epilogue.
// M=2 register blocking: wave computes 32 rows x 16 cols per gate; the two
// A-fragments amortize across all four gate B-matrices (8 wmma / 6 b128-pair
// loads per K step).
__global__ void gate_gemm_kernel(const __bf16* __restrict__ comb,
                                 const __bf16* __restrict__ wfb,
                                 const __bf16* __restrict__ wib,
                                 const __bf16* __restrict__ wob,
                                 const __bf16* __restrict__ wCb,
                                 const float* __restrict__ bf_,
                                 const float* __restrict__ bi_,
                                 const float* __restrict__ bo_,
                                 const float* __restrict__ bC_,
                                 const float* __restrict__ cell_in,
                                 const float* __restrict__ nw2,
                                 float* __restrict__ hid_out,
                                 float* __restrict__ cell_out) {
    int lane = threadIdx.x & 31, wave = threadIdx.x >> 5;
    int m0 = (blockIdx.y * 8 + wave) * 32;
    int n0 = blockIdx.x * 16;
    const __bf16* W[4] = {wfb, wib, wob, wCb};
    v8f acc[4][2] = {{ZERO8, ZERO8}, {ZERO8, ZERO8}, {ZERO8, ZERO8}, {ZERO8, ZERO8}};
    for (int k0 = 0; k0 < CN; k0 += 32) {
        v16bf a0 = load_a_frag(comb + (size_t)(m0)      * CN + k0, CN, lane);
        v16bf a1 = load_a_frag(comb + (size_t)(m0 + 16) * CN + k0, CN, lane);
#pragma unroll
        for (int g = 0; g < 4; ++g) {
            v16bf b = load_b_frag(W[g] + (size_t)n0 * CN + k0, CN, lane);
            acc[g][0] = wmma_bf16(a0, b, acc[g][0]);
            acc[g][1] = wmma_bf16(a1, b, acc[g][1]);
        }
    }
    int col = n0 + (lane & 15), mb = (lane >> 4) * 8;
    float bF = bf_[col], bI = bi_[col], bO = bo_[col], bC2 = bC_[col];
    float nwv = nw2[col];
#pragma unroll
    for (int mt = 0; mt < 2; ++mt) {
#pragma unroll
        for (int r = 0; r < 8; ++r) {
            size_t idx = (size_t)(m0 + mt * 16 + mb + r) * NDIM + col;
            float fv = sigm(acc[0][mt][r] + bF);
            float iv = sigm(acc[1][mt][r] + bI);
            float ov = sigm(acc[2][mt][r] + bO);
            float Cv = tanhf(acc[3][mt][r] + bC2);
            float cell = fv * (cell_in[idx] * nwv) + iv * Cv;
            cell_out[idx] = cell;
            hid_out[idx]  = ov * tanhf(cell);
        }
    }
}

// Recurrent path: four GEMMs over rcombined, fused LSTM epilogue, M=2 blocking.
__global__ void rgate_gemm_kernel(const __bf16* __restrict__ rcomb,
                                  const __bf16* __restrict__ wfb,
                                  const __bf16* __restrict__ wib,
                                  const __bf16* __restrict__ wob,
                                  const __bf16* __restrict__ wCb,
                                  const float* __restrict__ bf_,
                                  const float* __restrict__ bi_,
                                  const float* __restrict__ bo_,
                                  const float* __restrict__ bC_,
                                  const float* __restrict__ rcell_in,
                                  float* __restrict__ rhid_out,
                                  float* __restrict__ rcell_out) {
    int lane = threadIdx.x & 31, wave = threadIdx.x >> 5;
    int m0 = (blockIdx.y * 8 + wave) * 32;
    int n0 = blockIdx.x * 16;
    const __bf16* W[4] = {wfb, wib, wob, wCb};
    v8f acc[4][2] = {{ZERO8, ZERO8}, {ZERO8, ZERO8}, {ZERO8, ZERO8}, {ZERO8, ZERO8}};
    for (int k0 = 0; k0 < RN; k0 += 32) {
        v16bf a0 = load_a_frag(rcomb + (size_t)(m0)      * RN + k0, RN, lane);
        v16bf a1 = load_a_frag(rcomb + (size_t)(m0 + 16) * RN + k0, RN, lane);
#pragma unroll
        for (int g = 0; g < 4; ++g) {
            v16bf b = load_b_frag(W[g] + (size_t)n0 * RN + k0, RN, lane);
            acc[g][0] = wmma_bf16(a0, b, acc[g][0]);
            acc[g][1] = wmma_bf16(a1, b, acc[g][1]);
        }
    }
    int col = n0 + (lane & 15), mb = (lane >> 4) * 8;
    float bF = bf_[col], bI = bi_[col], bO = bo_[col], bC2 = bC_[col];
#pragma unroll
    for (int mt = 0; mt < 2; ++mt) {
#pragma unroll
        for (int r = 0; r < 8; ++r) {
            size_t idx = (size_t)(m0 + mt * 16 + mb + r) * NDIM + col;
            float fv = sigm(acc[0][mt][r] + bF);
            float iv = sigm(acc[1][mt][r] + bI);
            float ov = sigm(acc[2][mt][r] + bO);
            float Cv = tanhf(acc[3][mt][r] + bC2);
            float cell = fv * rcell_in[idx] + iv * Cv;
            rcell_out[idx] = cell;
            rhid_out[idx]  = ov * tanhf(cell);
        }
    }
}

// ---------------------------------------------------------------------------
// Deterministic two-stage variance reduction (f64 accumulation)
// ---------------------------------------------------------------------------
__global__ void var_partial_kernel(const float* __restrict__ x, long long n,
                                   double* __restrict__ part) {
    __shared__ double s1[256], s2[256];
    int tid = threadIdx.x;
    double s = 0.0, sq = 0.0;
    for (long long i = (long long)blockIdx.x * 256 + tid; i < n;
         i += (long long)gridDim.x * 256) {
        double v = (double)x[i];
        s += v; sq += v * v;
    }
    s1[tid] = s; s2[tid] = sq; __syncthreads();
    for (int o = 128; o > 0; o >>= 1) {
        if (tid < o) { s1[tid] += s1[tid + o]; s2[tid] += s2[tid + o]; }
        __syncthreads();
    }
    if (tid == 0) { part[2 * blockIdx.x] = s1[0]; part[2 * blockIdx.x + 1] = s2[0]; }
}

__global__ void var_final_kernel(const double* __restrict__ part, int nblocks,
                                 long long n, float* __restrict__ out) {
    __shared__ double s1[256], s2[256];
    int tid = threadIdx.x;
    double s = 0.0, sq = 0.0;
    for (int i = tid; i < nblocks; i += 256) { s += part[2 * i]; sq += part[2 * i + 1]; }
    s1[tid] = s; s2[tid] = sq; __syncthreads();
    for (int o = 128; o > 0; o >>= 1) {
        if (tid < o) { s1[tid] += s1[tid + o]; s2[tid] += s2[tid + o]; }
        __syncthreads();
    }
    if (tid == 0) {
        double mu = s1[0] / (double)n;
        out[0] = (float)((s2[0] - mu * s1[0]) / (double)(n - 1));  // ddof=1
    }
}

// pred = (Hidden*var1*c + rHidden*var2) / (var1 + var2*c)
__global__ void pred_kernel(const float* __restrict__ hid,
                            const float* __restrict__ rhid,
                            const float* __restrict__ varbuf,
                            const float* __restrict__ c,
                            float* __restrict__ pred) {
    float v1 = varbuf[0], v2 = varbuf[1], cv = c[0];
    float inv = 1.0f / (v1 + v2 * cv);
    for (int i = blockIdx.x * blockDim.x + threadIdx.x; i < BATCH * NDIM;
         i += gridDim.x * blockDim.x)
        pred[i] = (hid[i] * v1 * cv + rhid[i] * v2) * inv;
}

// ---------------------------------------------------------------------------
// Host-side orchestration
// ---------------------------------------------------------------------------
extern "C" void kernel_launch(void* const* d_in, const int* in_sizes, int n_in,
                              void* d_out, int out_size, void* d_ws, size_t ws_size,
                              hipStream_t stream) {
    (void)in_sizes; (void)n_in; (void)out_size; (void)ws_size;
    const float* input  = (const float*)d_in[0];
    const float* Hid0   = (const float*)d_in[1];
    const float* Cell0  = (const float*)d_in[2];
    const float* rHid0  = (const float*)d_in[3];
    const float* rCell0 = (const float*)d_in[4];
    const float* A      = (const float*)d_in[5];
    const float* Wgc    = (const float*)d_in[6];
    const float* wf     = (const float*)d_in[7];
    const float* bf_    = (const float*)d_in[8];
    const float* wi     = (const float*)d_in[9];
    const float* bi_    = (const float*)d_in[10];
    const float* wo     = (const float*)d_in[11];
    const float* bo_    = (const float*)d_in[12];
    const float* wC     = (const float*)d_in[13];
    const float* bC_    = (const float*)d_in[14];
    const float* nwt    = (const float*)d_in[15];
    const float* rwf    = (const float*)d_in[16];
    const float* rbf    = (const float*)d_in[17];
    const float* rwi    = (const float*)d_in[18];
    const float* rbi    = (const float*)d_in[19];
    const float* rwo    = (const float*)d_in[20];
    const float* rbo    = (const float*)d_in[21];
    const float* rwC    = (const float*)d_in[22];
    const float* rbC    = (const float*)d_in[23];
    const float* cptr   = (const float*)d_in[24];

    float* out = (float*)d_out;
    const size_t BN = (size_t)BATCH * NDIM;
    float* hid_out   = out;                 // [B,N]
    float* cell_out  = out + BN;            // [B,N]
    float* gc_out    = out + 2 * BN;        // [B,3N]
    float* rhid_out  = out + 5 * BN;        // [B,N]
    float* rcell_out = out + 6 * BN;        // [B,N]
    float* pred_out  = out + 7 * BN;        // [B,N]

    // Workspace carve-out (f64 first, then f32, then bf16 — all 16B aligned)
    uint8_t* p = (uint8_t*)d_ws;
    double* partials = (double*)p;               p += 2048 * sizeof(double);
    float*  cs       = (float*)p;                p += NDIM * sizeof(float);
    float*  nw2      = (float*)p;                p += NDIM * sizeof(float);
    float*  varbuf   = (float*)p;                p += 4 * sizeof(float);
    p = (uint8_t*)(((uintptr_t)p + 15) & ~(uintptr_t)15);
    float*  Af       = (float*)p;                p += (size_t)KHOP * NDIM * NDIM * sizeof(float);
    __bf16* normAT   = (__bf16*)p;               p += (size_t)NDIM * NDIM * 2;
    __bf16* Abf      = (__bf16*)p;               p += (size_t)KHOP * NDIM * NDIM * 2;
    __bf16* filt     = (__bf16*)p;               p += (size_t)KHOP * NDIM * NDIM * 2;
    __bf16* wbf[4];
    for (int g = 0; g < 4; ++g) { wbf[g] = (__bf16*)p; p += (size_t)NDIM * CN * 2; }
    __bf16* rwbf[4];
    for (int g = 0; g < 4; ++g) { rwbf[g] = (__bf16*)p; p += (size_t)NDIM * RN * 2; }
    __bf16* comb  = (__bf16*)p;                  p += BN * (CN / NDIM) * 2;  // [B,4096]
    __bf16* rcomb = (__bf16*)p;                  p += BN * (RN / NDIM) * 2;  // [B,2048]

    // 1) weights -> bf16
    const float* wsrc[4]  = {wf, wi, wo, wC};
    const float* rwsrc[4] = {rwf, rwi, rwo, rwC};
    for (int g = 0; g < 4; ++g)
        convert_f32_bf16_kernel<<<2048, 256, 0, stream>>>(wsrc[g], wbf[g], NDIM * CN);
    for (int g = 0; g < 4; ++g)
        convert_f32_bf16_kernel<<<2048, 256, 0, stream>>>(rwsrc[g], rwbf[g], NDIM * RN);

    // 2) graph preprocessing: colsum, normA^T, A1, chain A2/A3, filters, nw2
    colsum_kernel<<<NDIM / 256, 256, 0, stream>>>(A, cs);
    build_a1_kernel<<<NDIM * NDIM / 256, 256, 0, stream>>>(A, cs, normAT, Af, Abf);
    const size_t NN = (size_t)NDIM * NDIM;
    chain_gemm_kernel<<<dim3(NDIM / 16, NDIM / 128), 256, 0, stream>>>(
        Abf, normAT, Af + NN, Abf + NN);
    chain_gemm_kernel<<<dim3(NDIM / 16, NDIM / 128), 256, 0, stream>>>(
        Abf + NN, normAT, Af + 2 * NN, Abf + 2 * NN);
    filt_kernel<<<KHOP * NDIM * NDIM / 256, 256, 0, stream>>>(Af, Wgc, filt);
    nw2_kernel<<<NDIM, 256, 0, stream>>>(Af + 2 * NN, nwt, nw2);

    // 3) pack activations to bf16
    pack_kernel<<<(int)(BN / 256), 256, 0, stream>>>(input, rHid0, Hid0, rcomb, comb);

    // 4) gc GEMMs, 32x32 per wave (writes gc f32 output + combined bf16)
    gc_gemm_kernel<<<dim3(NDIM / 32, BATCH / 256, KHOP), 256, 0, stream>>>(
        rcomb, filt, gc_out, comb);

    // 5) fused gate GEMMs + LSTM cell/hidden update (32 rows/wave, 4 gates)
    gate_gemm_kernel<<<dim3(NDIM / 16, BATCH / 256), 256, 0, stream>>>(
        comb, wbf[0], wbf[1], wbf[2], wbf[3], bf_, bi_, bo_, bC_,
        Cell0, nw2, hid_out, cell_out);

    // 6) fused recurrent gate GEMMs + LSTM update
    rgate_gemm_kernel<<<dim3(NDIM / 16, BATCH / 256), 256, 0, stream>>>(
        rcomb, rwbf[0], rwbf[1], rwbf[2], rwbf[3], rbf, rbi, rbo, rbC,
        rCell0, rhid_out, rcell_out);

    // 7) variances (deterministic two-stage f64 reduction)
    var_partial_kernel<<<1024, 256, 0, stream>>>(input, (long long)BN, partials);
    var_final_kernel<<<1, 256, 0, stream>>>(partials, 1024, (long long)BN, varbuf);
    var_partial_kernel<<<1024, 256, 0, stream>>>(gc_out, (long long)(3 * BN), partials);
    var_final_kernel<<<1, 256, 0, stream>>>(partials, 1024, (long long)(3 * BN), varbuf + 1);

    // 8) final blend
    pred_kernel<<<2048, 256, 0, stream>>>(hid_out, rhid_out, varbuf, cptr, pred_out);
}
